// FNO2D_16647293239700
// MI455X (gfx1250) — matrix operand order repaired and checked by reading, not compiled, not run on hardware
//
#include <hip/hip_runtime.h>
#include <hip/hip_bf16.h>
#include <stdint.h>
#include <math.h>

// FNO2D forward for MI455X (gfx1250, wave32, WMMA).
// Spectral conv implemented as truncated-DFT GEMMs on v_wmma_f32_16x16x32_f16
// (f16 operands, f32 accumulation). Workspace use ~152 MB.

typedef __attribute__((ext_vector_type(16))) _Float16 v16h;
typedef __attribute__((ext_vector_type(8)))  _Float16 v8h;
typedef __attribute__((ext_vector_type(8)))  float    v8f;

#define BATCH 8
#define HH 256
#define WW 256
#define CH 32
#define MODES 12
#define TWO_PI 6.28318530717958647692f

__device__ __forceinline__ float gelu_exact(float x){
  return 0.5f * x * (1.0f + erff(x * 0.70710678118654752f));
}

// A-fragment (16x32 f16): LDS tile stored [m][k] row-major, pitch in halves
// (multiple of 8 => 16B-aligned vector loads).
// lane<16: m=lane,  K = {b0..b0+7, b0+16..b0+23}, b0=0 ; lane>=16: b0=8.
__device__ __forceinline__ v16h frag_a(const _Float16* s, int row0, int pitch, int k0, int lane){
  int m  = row0 + (lane & 15);
  int b0 = (lane < 16) ? 0 : 8;
  const _Float16* p = s + m * pitch + k0 + b0;
  v8h lo = *(const v8h*)(p);
  v8h hi = *(const v8h*)(p + 16);
  v16h a;
#pragma unroll
  for (int i = 0; i < 8; ++i){ a[i] = lo[i]; a[i+8] = hi[i]; }
  return a;
}

// B-fragment (32x16 f16): LDS tile stored transposed [n][k] row-major.
// lane<16: n=lane, K = {0..7, 16..23} ; lane>=16: n=lane-16, K = {8..15, 24..31}.
__device__ __forceinline__ v16h frag_b(const _Float16* s, int col0, int pitch, int k0, int lane){
  int n  = col0 + (lane & 15);
  int ks = k0 + ((lane < 16) ? 0 : 8);
  const _Float16* p = s + n * pitch + ks;
  v8h lo = *(const v8h*)(p);
  v8h hi = *(const v8h*)(p + 16);
  v16h b;
#pragma unroll
  for (int i = 0; i < 8; ++i){ b[i] = lo[i]; b[i+8] = hi[i]; }
  return b;
}

// ---------------- table init ----------------
// Fw[32][256] f16: rows 0..11 = cos(2*pi*w*ky/256), rows 16..27 = -sin(...)  (rfft)
// Gw[256][32] f16: cols 0..11 = (c_k/256)cos, cols 16..27 = -(c_k/256)sin    (irfft, c_0=1 else 2)
// trig[256][2] f32: {cos, sin}(2*pi*t/256)
__global__ void fno_init_tables(_Float16* __restrict__ Fw, _Float16* __restrict__ Gw,
                                float* __restrict__ trig){
  int i = blockIdx.x * blockDim.x + threadIdx.x;
  if (i < 32*256){
    int m = i >> 8, w = i & 255;
    float v = 0.f;
    if (m < MODES)                       v =  cosf(TWO_PI * (float)((w*m) & 255) * (1.f/256.f));
    else if (m >= 16 && m < 16+MODES){ int j = m-16;
                                         v = -sinf(TWO_PI * (float)((w*j) & 255) * (1.f/256.f)); }
    Fw[i] = (_Float16)v;
  } else if (i < 32*256 + 256*32){
    int j = i - 32*256; int w = j >> 5, k = j & 31;
    float v = 0.f;
    if (k < MODES){ float c = (k==0)?1.f:2.f;
      v =  (c*(1.f/256.f)) * cosf(TWO_PI * (float)((w*k) & 255) * (1.f/256.f)); }
    else if (k >= 16 && k < 16+MODES){ int j2 = k-16; float c = (j2==0)?1.f:2.f;
      v = -(c*(1.f/256.f)) * sinf(TWO_PI * (float)((w*j2) & 255) * (1.f/256.f)); }
    Gw[j] = (_Float16)v;
  } else if (i < 32*256 + 256*32 + 256){
    int t = i - (32*256 + 256*32);
    trig[2*t]   = cosf(TWO_PI * (float)t * (1.f/256.f));
    trig[2*t+1] = sinf(TWO_PI * (float)t * (1.f/256.f));
  }
}

// ---------------- fc0: (B*H*W,3) @ (3,32) + b ----------------
__global__ void fno_fc0(const float* __restrict__ x, const float* __restrict__ w,
                        const float* __restrict__ b, float* __restrict__ h){
  int t = blockIdx.x * blockDim.x + threadIdx.x;   // p*32 + c
  int p = t >> 5, c = t & 31;
  float x0 = x[p*3+0], x1 = x[p*3+1], x2 = x[p*3+2];
  h[t] = x0*w[c] + x1*w[32+c] + x2*w[64+c] + b[c];
}

// ---------------- K2: DFT along W (WMMA) ----------------
// Per (b,y): Zw(24x32, padded to 32x32) = Fw(32x256) x X(256x32). K=256 in 8 steps.
// Zw layout: [b][y][reim(2)][ky(16)][c(32)]; pad ky rows 12..15 are exact zeros.
__global__ void __launch_bounds__(128) fno_dftw(const float* __restrict__ hin,
                                                const _Float16* __restrict__ Fw,
                                                float* __restrict__ Zw){
  __shared__ _Float16 sA[32*264];   // twiddles [m][k], pitch 264
  __shared__ _Float16 sB[32*264];   // activations transposed [c][w], pitch 264
  int by  = blockIdx.x;             // b*256 + y
  int tid = threadIdx.x;
  const float* row = hin + (size_t)by * (WW*CH);
  for (int i = tid; i < WW*CH; i += 128){
    int w = i >> 5, c = i & 31;
    sB[c*264 + w] = (_Float16)row[i];
  }
  const uint32_t* fw32 = (const uint32_t*)Fw;
  for (int i = tid; i < 32*128; i += 128){
    int m = i >> 7, kp = i & 127;
    *(uint32_t*)&sA[m*264 + kp*2] = fw32[i];
  }
  __syncthreads();
  int wv = tid >> 5, lane = tid & 31;
  int mt = wv >> 1, nt = wv & 1;
  v8f acc = {};
#pragma unroll
  for (int kb = 0; kb < 8; ++kb){
    v16h a = frag_a(sA, mt*16, 264, kb*32, lane);
    v16h b = frag_b(sB, nt*16, 264, kb*32, lane);
    acc = __builtin_amdgcn_wmma_f32_16x16x32_f16(false, a, false, b, (short)0, acc, false, false);
  }
  float* out = Zw + (size_t)by * (2*16*32);
  int n = nt*16 + (lane & 15);
#pragma unroll
  for (int r = 0; r < 8; ++r){
    int m = mt*16 + r + ((lane < 16) ? 0 : 8);   // m = reim*16 + ky
    out[m*32 + n] = acc[r];
  }
}

// ---------------- K3a: DFT along H (complex, VALU + trig table) ----------------
__global__ void fno_dfth(const float* __restrict__ Zw, const float* __restrict__ trig,
                         float* __restrict__ Zh){
  __shared__ float tc[256], ts[256];
  int tid = threadIdx.x;
  tc[tid] = trig[2*tid]; ts[tid] = trig[2*tid+1];
  __syncthreads();
  int t  = blockIdx.x * 256 + tid;           // over b*12*16*32
  int c  = t & 31;
  int ky = (t >> 5) & 15;
  int kx = (t >> 9) % 12;
  int b  = t / (12*16*32);
  if (b >= BATCH) return;
  float re = 0.f, im = 0.f;
  int ph = 0;
  const float* zb = Zw + ((size_t)b*256)*(2*16*32) + ky*32 + c;
  for (int h = 0; h < 256; ++h){
    float a  = zb[(size_t)h*1024];
    float bb = zb[(size_t)h*1024 + 512];
    float ct = tc[ph], st = ts[ph];
    re += a*ct + bb*st;          // (a+ib)*e^{-i theta}
    im += bb*ct - a*st;
    ph = (ph + kx) & 255;
  }
  Zh[(((b*2+0)*16 + kx)*16 + ky)*32 + c] = re;
  Zh[(((b*2+1)*16 + kx)*16 + ky)*32 + c] = im;
}

// ---------------- K3b: per-mode complex 32x32 channel mix ----------------
__global__ void fno_modemix(const float* __restrict__ Zh, const float* __restrict__ cw,
                            float* __restrict__ Z3){
  int t  = blockIdx.x * 256 + threadIdx.x;   // over b*12*12*32
  int o  = t & 31;
  int ky = (t >> 5) % 12;
  int kx = (t / (32*12)) % 12;
  int b  = t / (32*144);
  if (b >= BATCH) return;
  float re = 0.f, im = 0.f;
  const float* zr = Zh + (((b*2+0)*16 + kx)*16 + ky)*32;
  const float* zi = Zh + (((b*2+1)*16 + kx)*16 + ky)*32;
  for (int c = 0; c < 32; ++c){
    float a = zr[c], bb = zi[c];
    const float* wp = cw + ((((size_t)c*32 + o)*12 + kx)*12 + ky) * 2;
    float wr = wp[0], wi = wp[1];
    re += a*wr - bb*wi;
    im += a*wi + bb*wr;
  }
  Z3[(((b*2+0)*16 + kx)*16 + ky)*32 + o] = re;
  Z3[(((b*2+1)*16 + kx)*16 + ky)*32 + o] = im;
}

// ---------------- K3c: IDFT along H (complex, 1/256), pad ky rows zeroed ----------------
__global__ void fno_idfth(const float* __restrict__ Z3, const float* __restrict__ trig,
                          float* __restrict__ Yh){
  __shared__ float tc[256], ts[256];
  int tid = threadIdx.x;
  tc[tid] = trig[2*tid]; ts[tid] = trig[2*tid+1];
  __syncthreads();
  int t  = blockIdx.x * 256 + tid;           // over b*256*16*32
  int o  = t & 31;
  int ky = (t >> 5) & 15;
  int h  = (t >> 9) & 255;
  int b  = t >> 17;
  size_t oidx = (((size_t)(b*256 + h)*2 + 0)*16 + ky)*32 + o;
  if (ky >= MODES){ Yh[oidx] = 0.f; Yh[oidx + 512] = 0.f; return; }
  float re = 0.f, im = 0.f;
  int ph = 0;
  for (int kx = 0; kx < MODES; ++kx){
    float a  = Z3[(((b*2+0)*16 + kx)*16 + ky)*32 + o];
    float bb = Z3[(((b*2+1)*16 + kx)*16 + ky)*32 + o];
    float ct = tc[ph], st = ts[ph];
    re += a*ct - bb*st;          // (a+ib)*e^{+i theta}
    im += a*st + bb*ct;
    ph = (ph + h) & 255;
  }
  Yh[oidx]       = re * (1.f/256.f);
  Yh[oidx + 512] = im * (1.f/256.f);
}

// ---------------- K4: IDFT along W + pointwise conv + bias + GELU (WMMA, fused) ----------------
// Per (b,y): out(256x32) = Gw(256x32) x [ReYh;ImYh](32x32)  +  X(256x32) x Wl(32x32) + b
__global__ void __launch_bounds__(128) fno_idftw_pw(const float* __restrict__ hin,
    const float* __restrict__ Yh, const _Float16* __restrict__ Gw,
    const float* __restrict__ wl, const float* __restrict__ bl,
    float* __restrict__ hout, int do_gelu){
  __shared__ _Float16 sA1[256*40];  // Gw [w][k]
  __shared__ _Float16 sA2[256*40];  // X  [w][c]
  __shared__ _Float16 sB1[32*40];   // Yh^T [o][k]   (k<16 -> Re[ky=k], k>=16 -> Im[ky=k-16])
  __shared__ _Float16 sB2[32*40];   // Wl^T [o][c]
  int by  = blockIdx.x;
  int tid = threadIdx.x;
  const float* row  = hin + (size_t)by * (WW*CH);
  const float* yrow = Yh  + (size_t)by * (2*16*32);
  const uint32_t* gw32 = (const uint32_t*)Gw;
  for (int i = tid; i < 256*16; i += 128){
    int w = i >> 4, kp = i & 15;
    *(uint32_t*)&sA1[w*40 + kp*2] = gw32[i];
  }
  for (int i = tid; i < 256*32; i += 128){
    int w = i >> 5, c = i & 31;
    sA2[w*40 + c] = (_Float16)row[i];
  }
  for (int i = tid; i < 1024; i += 128){
    int k = i >> 5, n = i & 31;
    float v = (k < 16) ? yrow[k*32 + n] : yrow[512 + (k-16)*32 + n];
    sB1[n*40 + k] = (_Float16)v;
  }
  for (int i = tid; i < 1024; i += 128){
    int c = i >> 5, o = i & 31;
    sB2[o*40 + c] = (_Float16)wl[c*32 + o];
  }
  __syncthreads();
  int wv = tid >> 5, lane = tid & 31;
  float* out = hout + (size_t)by * (WW*CH);
#pragma unroll
  for (int mi = 0; mi < 4; ++mi){
    int mt = wv*4 + mi;
    v16h a1 = frag_a(sA1, mt*16, 40, 0, lane);
    v16h a2 = frag_a(sA2, mt*16, 40, 0, lane);
#pragma unroll
    for (int nt = 0; nt < 2; ++nt){
      v8f acc = {};
      v16h b1 = frag_b(sB1, nt*16, 40, 0, lane);
      acc = __builtin_amdgcn_wmma_f32_16x16x32_f16(false, a1, false, b1, (short)0, acc, false, false);
      v16h b2 = frag_b(sB2, nt*16, 40, 0, lane);
      acc = __builtin_amdgcn_wmma_f32_16x16x32_f16(false, a2, false, b2, (short)0, acc, false, false);
      int o = nt*16 + (lane & 15);
      float bias = bl[o];
#pragma unroll
      for (int r = 0; r < 8; ++r){
        int w = mt*16 + r + ((lane < 16) ? 0 : 8);
        float v = acc[r] + bias;
        if (do_gelu) v = gelu_exact(v);
        out[w*32 + o] = v;
      }
    }
  }
}

// ---------------- K5: fc1(32->128)+GELU fused with fc2(128->1) (WMMA + shuffle reduce) ----------------
__global__ void __launch_bounds__(256) fno_head(const float* __restrict__ hin,
    const float* __restrict__ w1, const float* __restrict__ b1,
    const float* __restrict__ w2, const float* __restrict__ b2,
    float* __restrict__ out){
  __shared__ _Float16 sA[128*40];   // pixels [m][c]
  __shared__ _Float16 sB[128*40];   // fc1_w^T [n][c]
  __shared__ float sb1[128], sw2[128];
  int tid = threadIdx.x;
  size_t p0 = (size_t)blockIdx.x * 128;
  for (int i = tid; i < 128*32; i += 256){
    int m = i >> 5, c = i & 31;
    sA[m*40 + c] = (_Float16)hin[(p0 + m)*32 + c];
  }
  for (int i = tid; i < 32*128; i += 256){
    int c = i >> 7, n = i & 127;
    sB[n*40 + c] = (_Float16)w1[i];
  }
  if (tid < 128){ sb1[tid] = b1[tid]; sw2[tid] = w2[tid]; }
  __syncthreads();
  int wv = tid >> 5, lane = tid & 31;
  int mt = wv;
  float s[8];
#pragma unroll
  for (int r = 0; r < 8; ++r) s[r] = 0.f;
  v16h a = frag_a(sA, mt*16, 40, 0, lane);
#pragma unroll
  for (int nt = 0; nt < 8; ++nt){
    v16h b = frag_b(sB, nt*16, 40, 0, lane);
    v8f acc = {};
    acc = __builtin_amdgcn_wmma_f32_16x16x32_f16(false, a, false, b, (short)0, acc, false, false);
    int nb = nt*16 + (lane & 15);
    float bb = sb1[nb], ww = sw2[nb];
#pragma unroll
    for (int r = 0; r < 8; ++r){
      float g = gelu_exact(acc[r] + bb);
      s[r] += g * ww;
    }
  }
#pragma unroll
  for (int r = 0; r < 8; ++r){
    s[r] += __shfl_xor(s[r], 1, 32);
    s[r] += __shfl_xor(s[r], 2, 32);
    s[r] += __shfl_xor(s[r], 4, 32);
    s[r] += __shfl_xor(s[r], 8, 32);
  }
  if ((lane & 15) == 0){
    float bias2 = b2[0];
#pragma unroll
    for (int r = 0; r < 8; ++r){
      int m = mt*16 + r + ((lane < 16) ? 0 : 8);
      out[p0 + m] = s[r] + bias2;
    }
  }
}

extern "C" void kernel_launch(void* const* d_in, const int* in_sizes, int n_in,
                              void* d_out, int out_size, void* d_ws, size_t ws_size,
                              hipStream_t stream){
  (void)in_sizes; (void)n_in; (void)out_size; (void)ws_size;
  const float* x      = (const float*)d_in[0];
  const float* fc0_w  = (const float*)d_in[1];
  const float* fc0_b  = (const float*)d_in[2];
  const float* conv_w = (const float*)d_in[3];
  const float* w_w    = (const float*)d_in[4];
  const float* w_b    = (const float*)d_in[5];
  const float* fc1_w  = (const float*)d_in[6];
  const float* fc1_b  = (const float*)d_in[7];
  const float* fc2_w  = (const float*)d_in[8];
  const float* fc2_b  = (const float*)d_in[9];
  float* out = (float*)d_out;

  // workspace carve (~152 MB total)
  char* p = (char*)d_ws;
  float*    hA   = (float*)p;    p += (size_t)BATCH*HH*WW*CH*4;     // 67.1 MB
  float*    hB   = (float*)p;    p += (size_t)BATCH*HH*WW*CH*4;     // 67.1 MB
  float*    Zw   = (float*)p;    p += (size_t)BATCH*HH*2*16*32*4;   // 8.4 MB
  float*    Yh   = (float*)p;    p += (size_t)BATCH*HH*2*16*32*4;   // 8.4 MB
  float*    Zh   = (float*)p;    p += (size_t)BATCH*2*16*16*32*4;   // 0.5 MB
  float*    Z3   = (float*)p;    p += (size_t)BATCH*2*16*16*32*4;   // 0.5 MB
  _Float16* Fw   = (_Float16*)p; p += 32*256*2;
  _Float16* Gw   = (_Float16*)p; p += 256*32*2;
  float*    trig = (float*)p;    p += 256*2*4;

  fno_init_tables<<<65, 256, 0, stream>>>(Fw, Gw, trig);
  fno_fc0<<<65536, 256, 0, stream>>>(x, fc0_w, fc0_b, hA);

  float* cur = hA; float* nxt = hB;
  for (int l = 0; l < 4; ++l){
    fno_dftw<<<BATCH*HH, 128, 0, stream>>>(cur, Fw, Zw);
    fno_dfth<<<192, 256, 0, stream>>>(Zw, trig, Zh);
    fno_modemix<<<144, 256, 0, stream>>>(Zh, conv_w + (size_t)l*32*32*12*12*2, Z3);
    fno_idfth<<<4096, 256, 0, stream>>>(Z3, trig, Yh);
    fno_idftw_pw<<<BATCH*HH, 128, 0, stream>>>(cur, Yh, Gw,
        w_w + (size_t)l*32*32, w_b + (size_t)l*32, nxt, (l < 3) ? 1 : 0);
    float* t = cur; cur = nxt; nxt = t;
  }
  fno_head<<<4096, 256, 0, stream>>>(cur, fc1_w, fc1_b, fc2_w, fc2_b, out);
}
